// CrossAttentionLayer_86861418594526
// MI455X (gfx1250) — matrix operand order repaired and checked by reading, not compiled, not run on hardware
//
#include <hip/hip_runtime.h>

// ---------------------------------------------------------------------------
// Types (trivial ext-vector types so they can live in unions)
// ---------------------------------------------------------------------------
typedef __bf16 bf16;
typedef __attribute__((ext_vector_type(16))) __bf16 v16bf;
typedef __attribute__((ext_vector_type(8)))  float  v8f;
typedef __attribute__((ext_vector_type(4)))  unsigned int u32x4;
typedef __attribute__((ext_vector_type(2)))  unsigned int u32x2;
typedef __attribute__((ext_vector_type(4)))  float f32x4;
typedef __attribute__((ext_vector_type(4)))  int   i32x4;

// Address-space-qualified int4 pointers for the async global->LDS builtin
typedef __attribute__((address_space(1))) i32x4 gi32x4;   // global
typedef __attribute__((address_space(3))) i32x4 li32x4;   // LDS

union FragU { u32x4 u[2]; v16bf v; };

#if __has_builtin(__builtin_amdgcn_global_load_async_to_lds_b128) && \
    __has_builtin(__builtin_amdgcn_s_wait_asynccnt)
#define USE_ASYNC_LDS 1
#else
#define USE_ASYNC_LDS 0
#endif

__device__ __forceinline__ void wait_async0() {
#if USE_ASYNC_LDS
  __builtin_amdgcn_s_wait_asynccnt(0);
#else
  asm volatile("s_wait_asynccnt 0x0" ::: "memory");
#endif
}

__device__ __forceinline__ v8f wmma_bf16(v16bf a, v16bf b, v8f c) {
  // D = A(16x32) * B(32x16) + C, f32 accumulate
  return __builtin_amdgcn_wmma_f32_16x16x32_bf16(false, a, false, b,
                                                 (short)0, c, false, false);
}

// A-frag (16x32 bf16): lane m(0-15) -> row m, K 0..15; lanes 16-31 -> K 16..31
__device__ __forceinline__ v16bf lds_a_frag(const bf16* base, int ld) {
  const int lane = threadIdx.x & 31;
  const bf16* p = base + (lane & 15) * ld + ((lane >> 4) << 4);
  FragU f;
  f.u[0] = *(const u32x4*)(p);
  f.u[1] = *(const u32x4*)(p + 8);
  return f.v;
}

// B-frag (32x16 bf16): lane n(0-15) -> col n, K 0..15; lanes 16-31 -> K 16..31
__device__ __forceinline__ v16bf lds_b_frag(const bf16* base, int ld) {
  const int lane = threadIdx.x & 31;
  const bf16* p = base + (lane & 15) * ld + ((lane >> 4) << 4);
  FragU f;
  f.u[0] = *(const u32x4*)(p);
  f.u[1] = *(const u32x4*)(p + 8);
  return f.v;
}

// ---------------------------------------------------------------------------
// fp32 -> bf16 conversion pass (one float4 -> packed 4x bf16 per thread-iter)
// ---------------------------------------------------------------------------
__global__ __launch_bounds__(256)
void convert_f32_bf16_kernel(const float* __restrict__ src,
                             bf16* __restrict__ dst, int n4)
{
  const int i = blockIdx.x * 256 + threadIdx.x;
  if (i < n4) {
    f32x4 a = *(const f32x4*)(src + (size_t)i * 4);
    union { bf16 h[4]; u32x2 u; } t;
    t.h[0] = (bf16)a.x; t.h[1] = (bf16)a.y;
    t.h[2] = (bf16)a.z; t.h[3] = (bf16)a.w;
    *(u32x2*)(dst + (size_t)i * 4) = t.u;
  }
}

// ---------------------------------------------------------------------------
// GEMM: out = alpha * (A @ W^T + bias),  A:(M,K) bf16, W:(N,K) bf16
//   MODE 0: bf16 out row-major (M,N)
//   MODE 1: bf16 out transposed (N,M)          (for V^T)
//   MODE 2: fp32 out row-major (M,N)           (final projection)
// 128x128 block tile, 256 threads (8 waves), wave tile 64x32, K-step 32,
// double-buffered LDS fed by GLOBAL_LOAD_ASYNC_TO_LDS_B128 (ASYNCcnt).
// ---------------------------------------------------------------------------
#define TILE_K     32
#define LDS_STRIDE 48   // bf16 elements; 96B rows keep b128 chunks 16B aligned

template<int MODE>
__global__ __launch_bounds__(256, 2)
void gemm_xwT_kernel(const bf16* __restrict__ A, const bf16* __restrict__ W,
                     const float* __restrict__ bias, float alpha,
                     bf16* __restrict__ outb, float* __restrict__ outf,
                     int M, int N, int K)
{
  __shared__ bf16 sA[2][128][LDS_STRIDE];
  __shared__ bf16 sB[2][128][LDS_STRIDE];

  const int tx = threadIdx.x;
  const int m0 = blockIdx.y * 128;
  const int n0 = blockIdx.x * 128;

  // Stage loader: 128x32 bf16 tile = 512 x 16B chunks; 2 chunks/thread/tile.
  auto load_stage = [&](int kk, int buf) {
#pragma unroll
    for (int i = 0; i < 2; ++i) {
      const int ch  = tx + (i << 8);     // 0..511
      const int row = ch >> 2;           // 0..127
      const int col = (ch & 3) << 3;     // 0,8,16,24 (bf16 elems, 16B chunks)
      const bf16* ga = A + (size_t)(m0 + row) * K + kk + col;
      const bf16* gw = W + (size_t)(n0 + row) * K + kk + col;
      bf16* la = &sA[buf][row][col];
      bf16* lb = &sB[buf][row][col];
#if USE_ASYNC_LDS
      __builtin_amdgcn_global_load_async_to_lds_b128((gi32x4*)ga, (li32x4*)la, 0, 0);
      __builtin_amdgcn_global_load_async_to_lds_b128((gi32x4*)gw, (li32x4*)lb, 0, 0);
#else
      *(u32x4*)la = *(const u32x4*)ga;
      *(u32x4*)lb = *(const u32x4*)gw;
#endif
    }
  };

  const int wave = tx >> 5;
  const int wm = (wave & 1) * 64;    // 2 wave-rows of 64
  const int wn = (wave >> 1) * 32;   // 4 wave-cols of 32

  const v8f vzero = {0.f,0.f,0.f,0.f,0.f,0.f,0.f,0.f};
  v8f acc[4][2];
#pragma unroll
  for (int i = 0; i < 4; ++i) { acc[i][0] = vzero; acc[i][1] = vzero; }

  load_stage(0, 0);
  int buf = 0;
  for (int kk = 0; kk < K; kk += TILE_K, buf ^= 1) {
#if USE_ASYNC_LDS
    wait_async0();                     // my async copies for 'buf' landed
#endif
    __syncthreads();                   // everyone's copies for 'buf' visible
    if (kk + TILE_K < K) load_stage(kk + TILE_K, buf ^ 1);
    v16bf bfr0 = lds_b_frag(&sB[buf][wn][0],      LDS_STRIDE);
    v16bf bfr1 = lds_b_frag(&sB[buf][wn + 16][0], LDS_STRIDE);
#pragma unroll
    for (int i = 0; i < 4; ++i) {
      v16bf af = lds_a_frag(&sA[buf][wm + i * 16][0], LDS_STRIDE);
      acc[i][0] = wmma_bf16(af, bfr0, acc[i][0]);
      acc[i][1] = wmma_bf16(af, bfr1, acc[i][1]);
    }
  }

  // Epilogue. C-layout: VGPR r -> (row r, lanes 0-15) / (row 8+r, lanes 16-31)
  const int lane = tx & 31;
  const int cl = lane & 15;
  const int hi = lane >> 4;
#pragma unroll
  for (int j = 0; j < 2; ++j) {
    const int colg = n0 + wn + j * 16 + cl;
    const float bv = bias[colg];
#pragma unroll
    for (int i = 0; i < 4; ++i) {
      const int rowb = m0 + wm + i * 16;
      if (MODE == 1) {
        union { bf16 h[8]; u32x4 u; } t;
#pragma unroll
        for (int r = 0; r < 8; ++r)
          t.h[r] = (bf16)(alpha * (acc[i][j][r] + bv));
        *(u32x4*)(outb + (size_t)colg * M + rowb + hi * 8) = t.u;
      } else {
#pragma unroll
        for (int r = 0; r < 8; ++r) {
          const int row = rowb + r + hi * 8;
          const float val = alpha * (acc[i][j][r] + bv);
          if (MODE == 0) outb[(size_t)row * N + colg] = (bf16)val;
          else           outf[(size_t)row * N + colg] = val;
        }
      }
    }
  }
}

// ---------------------------------------------------------------------------
// Flash attention: one block per (b,h); 4 waves, each owns 16 query rows.
// q:  (B,NQ,D) bf16, pre-scaled by HD^-0.5
// k:  (B,S,D)  bf16                (row s contiguous in d -> B-frag friendly)
// vt: (D,B*S)  bf16 = V^T          (row hd contiguous in s -> B-frag friendly)
// ao: (B,NQ,D) bf16
// ---------------------------------------------------------------------------
__global__ __launch_bounds__(128)
void attention_kernel(const bf16* __restrict__ q, const bf16* __restrict__ k,
                      const bf16* __restrict__ vt, bf16* __restrict__ ao)
{
  constexpr int S  = 4096;
  constexpr int D  = 1024;
  constexpr int BS = 8 * S;     // V^T leading dimension

  __shared__ bf16 pbuf[4][16][32];   // per-wave P-transpose staging

  const int bh = blockIdx.x;
  const int b = bh >> 4, h = bh & 15;
  const int wave = threadIdx.x >> 5;
  const int lane = threadIdx.x & 31;
  const int cl = lane & 15, hi = lane >> 4;

  // Resident Q A-fragments (16 rows x 64 k, two 16x32 frags)
  const bf16* qrow = q + (size_t)(b * 64 + wave * 16 + cl) * D + h * 64;
  v16bf qa[2];
#pragma unroll
  for (int t = 0; t < 2; ++t) {
    FragU f;
    const bf16* p = qrow + t * 32 + hi * 16;
    f.u[0] = *(const u32x4*)(p);
    f.u[1] = *(const u32x4*)(p + 8);
    qa[t] = f.v;
  }

  const v8f vzero = {0.f,0.f,0.f,0.f,0.f,0.f,0.f,0.f};
  v8f oacc[4];
#pragma unroll
  for (int t = 0; t < 4; ++t) oacc[t] = vzero;
  float mrun[8], lrun[8];
#pragma unroll
  for (int r = 0; r < 8; ++r) { mrun[r] = -3.0e38f; lrun[r] = 0.f; }

  const bf16* kbase = k  + (size_t)b * S * D + h * 64;
  const bf16* vbase = vt + (size_t)(h * 64) * BS + (size_t)b * S;

  for (int st = 0; st < S; st += 32) {
    // ---- scores: two 16x16 tiles, reduction over HD=64 (2 WMMA each) ----
    v8f sc[2];
#pragma unroll
    for (int ct = 0; ct < 2; ++ct) {
      const bf16* krow = kbase + (size_t)(st + ct * 16 + cl) * D;
      FragU f0, f1;
      f0.u[0] = *(const u32x4*)(krow + hi * 16);
      f0.u[1] = *(const u32x4*)(krow + hi * 16 + 8);
      f1.u[0] = *(const u32x4*)(krow + 32 + hi * 16);
      f1.u[1] = *(const u32x4*)(krow + 32 + hi * 16 + 8);
      sc[ct] = wmma_bf16(qa[0], f0.v, vzero);
      sc[ct] = wmma_bf16(qa[1], f1.v, sc[ct]);
    }
    // ---- online softmax (row reductions across 16-lane groups) ----
    float corr[8];
#pragma unroll
    for (int r = 0; r < 8; ++r) {
      float t = fmaxf(sc[0][r], sc[1][r]);
      t = fmaxf(t, __shfl_xor(t, 8, 16));
      t = fmaxf(t, __shfl_xor(t, 4, 16));
      t = fmaxf(t, __shfl_xor(t, 2, 16));
      t = fmaxf(t, __shfl_xor(t, 1, 16));
      const float mn = fmaxf(mrun[r], t);
      corr[r] = __expf(mrun[r] - mn);
      mrun[r] = mn;
      const float p0 = __expf(sc[0][r] - mn);
      const float p1 = __expf(sc[1][r] - mn);
      sc[0][r] = p0; sc[1][r] = p1;
      float rs = p0 + p1;
      rs += __shfl_xor(rs, 8, 16);
      rs += __shfl_xor(rs, 4, 16);
      rs += __shfl_xor(rs, 2, 16);
      rs += __shfl_xor(rs, 1, 16);
      lrun[r] = lrun[r] * corr[r] + rs;
    }
#pragma unroll
    for (int t = 0; t < 4; ++t)
#pragma unroll
      for (int r = 0; r < 8; ++r) oacc[t][r] *= corr[r];

    // ---- transpose P (C-layout -> A-layout) through per-wave LDS ----
    bf16* pw = &pbuf[wave][0][0];
#pragma unroll
    for (int r = 0; r < 8; ++r) {
      pw[(r + hi * 8) * 32 + cl]      = (bf16)sc[0][r];
      pw[(r + hi * 8) * 32 + 16 + cl] = (bf16)sc[1][r];
    }
    asm volatile("s_wait_dscnt 0x0" ::: "memory");   // same-wave LDS RAW
    FragU pf;
    const bf16* pp = pw + cl * 32 + hi * 16;
    pf.u[0] = *(const u32x4*)(pp);
    pf.u[1] = *(const u32x4*)(pp + 8);
    const v16bf pa = pf.v;

    // ---- O += P(16x32) x V(32x64) via V^T B-frags ----
#pragma unroll
    for (int t = 0; t < 4; ++t) {
      const bf16* vrow = vbase + (size_t)(t * 16 + cl) * BS + st + hi * 16;
      FragU vf;
      vf.u[0] = *(const u32x4*)(vrow);
      vf.u[1] = *(const u32x4*)(vrow + 8);
      oacc[t] = wmma_bf16(pa, vf.v, oacc[t]);
    }
  }

  // ---- normalize and write (B,NQ,H,HD) bf16 ----
  bf16* orow = ao + (size_t)(b * 64 + wave * 16) * D + h * 64;
  float inv[8];
#pragma unroll
  for (int r = 0; r < 8; ++r) inv[r] = 1.0f / lrun[r];
#pragma unroll
  for (int t = 0; t < 4; ++t)
#pragma unroll
    for (int r = 0; r < 8; ++r)
      orow[(size_t)(r + hi * 8) * D + t * 16 + cl] = (bf16)(oacc[t][r] * inv[r]);
}

// ---------------------------------------------------------------------------
// Launch: bf16 casts -> Q/K/V projections -> attention -> output projection
// ---------------------------------------------------------------------------
extern "C" void kernel_launch(void* const* d_in, const int* in_sizes, int n_in,
                              void* d_out, int out_size, void* d_ws, size_t ws_size,
                              hipStream_t stream)
{
  const float* queries = (const float*)d_in[0];
  const float* context = (const float*)d_in[1];
  const float* wq = (const float*)d_in[2];
  const float* bq = (const float*)d_in[3];
  const float* wk = (const float*)d_in[4];
  const float* bk = (const float*)d_in[5];
  const float* wv = (const float*)d_in[6];
  const float* bv = (const float*)d_in[7];
  const float* wo = (const float*)d_in[8];
  const float* bo = (const float*)d_in[9];
  float* out = (float*)d_out;

  const int B = 8, NQ = 64, S = 4096, D = 1024;
  const int MQ = B * NQ;    // 512
  const int MK = B * S;     // 32768
  const float scale = 0.125f;   // HD^-0.5, folded into Q projection

  // Workspace (bf16): inputs cast once, then WMMA-ready tensors
  bf16* ctxb  = (bf16*)d_ws;                 // (MK, D)
  bf16* qinb  = ctxb  + (size_t)MK * D;      // (MQ, D)
  bf16* wqb   = qinb  + (size_t)MQ * D;      // (D, D)
  bf16* wkb   = wqb   + (size_t)D * D;
  bf16* wvb   = wkb   + (size_t)D * D;
  bf16* wob   = wvb   + (size_t)D * D;
  bf16* qb    = wob   + (size_t)D * D;       // (MQ, D)  scaled Q
  bf16* kb    = qb    + (size_t)MQ * D;      // (MK, D)
  bf16* vtb   = kb    + (size_t)MK * D;      // (D, MK)  V^T
  bf16* attnb = vtb   + (size_t)MK * D;      // (MQ, D)

  dim3 blk(256);
  // fp32 -> bf16 casts
  {
    const int n4c = (MK * D) / 4, n4q = (MQ * D) / 4, n4w = (D * D) / 4;
    convert_f32_bf16_kernel<<<dim3((n4c + 255) / 256), blk, 0, stream>>>(context, ctxb, n4c);
    convert_f32_bf16_kernel<<<dim3((n4q + 255) / 256), blk, 0, stream>>>(queries, qinb, n4q);
    convert_f32_bf16_kernel<<<dim3((n4w + 255) / 256), blk, 0, stream>>>(wq, wqb, n4w);
    convert_f32_bf16_kernel<<<dim3((n4w + 255) / 256), blk, 0, stream>>>(wk, wkb, n4w);
    convert_f32_bf16_kernel<<<dim3((n4w + 255) / 256), blk, 0, stream>>>(wv, wvb, n4w);
    convert_f32_bf16_kernel<<<dim3((n4w + 255) / 256), blk, 0, stream>>>(wo, wob, n4w);
  }
  // projections
  gemm_xwT_kernel<0><<<dim3(D / 128, MQ / 128), blk, 0, stream>>>(
      qinb, wqb, bq, scale, qb, nullptr, MQ, D, D);
  gemm_xwT_kernel<0><<<dim3(D / 128, MK / 128), blk, 0, stream>>>(
      ctxb, wkb, bk, 1.0f, kb, nullptr, MK, D, D);
  gemm_xwT_kernel<1><<<dim3(D / 128, MK / 128), blk, 0, stream>>>(
      ctxb, wvb, bv, 1.0f, vtb, nullptr, MK, D, D);
  // attention
  attention_kernel<<<dim3(B * 16), dim3(128), 0, stream>>>(qb, kb, vtb, attnb);
  // output projection (fp32 out)
  gemm_xwT_kernel<2><<<dim3(D / 128, MQ / 128), blk, 0, stream>>>(
      attnb, wob, bo, 1.0f, nullptr, out, MQ, D, D);
}
